// TexualEmbeddingLayer_40767829574160
// MI455X (gfx1250) — compile-verified, matrix operand order checked by reading, not run on hardware
//
#include <hip/hip_runtime.h>

// ---------------------------------------------------------------------------
// Types for CDNA5 WMMA (wave32, bf16 -> f32 accumulate)
// ---------------------------------------------------------------------------
typedef __bf16 v16bf __attribute__((ext_vector_type(16)));
typedef __bf16 v8bf  __attribute__((ext_vector_type(8)));
typedef float  v8f   __attribute__((ext_vector_type(8)));

__device__ inline v8f wmma_bf16(v16bf a, v16bf b, v8f c) {
    // (neg_a, A, neg_b, B, c_mod, C, reuse_a, reuse_b)
    return __builtin_amdgcn_wmma_f32_16x16x32_bf16(false, a, false, b, (short)0, c, false, false);
}

// Load a 16x32 bf16 fragment (A-operand layout, or B-operand loaded from
// W[N,K] rows) from LDS. base points at (row0, kchunk0); stride in elements.
__device__ inline v16bf ldfrag(const __bf16* base, int stride, int lane) {
    int r  = lane & 15;
    int hv = (lane >> 4) << 3;
    const __bf16* q = base + r * stride + hv;
    union { v16bf v; v8bf h[2]; } u;
    u.h[0] = *(const v8bf*)(q);
    u.h[1] = *(const v8bf*)(q + 16);
    return u.v;
}

// gfx1250 async memory->LDS copy (ASYNCcnt tracked). lds = raw LDS byte
// address (dynamic-LDS base is 0 when the kernel has no static __shared__),
// g = per-lane global address; copies 16 bytes.
__device__ inline void async_ld_b128(unsigned lds, const __bf16* g) {
    asm volatile("global_load_async_to_lds_b128 %0, %1, off"
                 :: "v"(lds), "v"(g) : "memory");
}
__device__ inline void wait_async0() {
    asm volatile("s_wait_asynccnt 0" ::: "memory");
}

// ---------------------------------------------------------------------------
// Problem constants
// ---------------------------------------------------------------------------
#define BATCH 256
#define NT    512
#define DDIM  512
#define EDIM  1024
#define TDIM  1536
#define KSEL  153
#define KP    160            // padded k (10 tiles of 16)
#define MR    (BATCH * KP)   // 40960 rows, divisible by 128

// GEMM LDS layout (dynamic shared memory, byte offsets from 0):
//   stage s (s=0,1): As at s*20480, Ws at s*20480 + 10240  (128x40 bf16 each)
#define STG_BYTES   20480
#define WS_BYTEOFF  10240
#define GEMM_LDS    (2 * STG_BYTES)

// ---------------------------------------------------------------------------
// Kernel 0: init output to -inf-ish for max pooling
// ---------------------------------------------------------------------------
__global__ void init_out_kernel(float* out, int n) {
    int i = blockIdx.x * blockDim.x + threadIdx.x;
    if (i < n) out[i] = -3.4e38f;
}

// ---------------------------------------------------------------------------
// Kernel 1: per-batch EOS/argmax, length, masked attention row, top-153 select
// ---------------------------------------------------------------------------
__global__ __launch_bounds__(256) void topk_kernel(const int* __restrict__ text,
                                                   const float* __restrict__ atten,
                                                   int* __restrict__ idx,
                                                   int* __restrict__ lens) {
    int b = blockIdx.x, tid = threadIdx.x;
    __shared__ float row[NT];
    __shared__ float rv[256];
    __shared__ int   ri[256];
    __shared__ int   rn[256];
    __shared__ int   eos_s;

    const int* tb = text + b * NT;
    int v0 = tb[tid], v1 = tb[tid + 256];
    int myv, myi;
    if (v0 >= v1) { myv = v0; myi = tid; } else { myv = v1; myi = tid + 256; }
    rv[tid] = (float)myv; ri[tid] = myi;
    rn[tid] = (v0 != 0) + (v1 != 0);
    __syncthreads();
    for (int s = 128; s > 0; s >>= 1) {
        if (tid < s) {
            if (rv[tid + s] > rv[tid] || (rv[tid + s] == rv[tid] && ri[tid + s] < ri[tid])) {
                rv[tid] = rv[tid + s]; ri[tid] = ri[tid + s];
            }
            rn[tid] += rn[tid + s];
        }
        __syncthreads();
    }
    if (tid == 0) {
        eos_s = ri[0];
        int L = rn[0] - 2;
        L = L < KSEL ? L : KSEL;
        L = L < 1 ? 1 : L;
        lens[b] = L;
    }
    __syncthreads();
    int eos = eos_s;

    for (int j = tid; j < NT; j += 256) {
        float m = (tb[j] != 0) ? 1.f : 0.f;
        float r = atten[(size_t)b * NT * NT + (size_t)eos * NT + j];
        if (j == eos || j == 0) r = -1.f; else r = r * m;
        row[j] = r;
    }
    __syncthreads();

    for (int i = 0; i < KSEL; i++) {
        float a0 = row[tid], a1 = row[tid + 256];
        float bv; int bi;
        if (a0 >= a1) { bv = a0; bi = tid; } else { bv = a1; bi = tid + 256; }
        rv[tid] = bv; ri[tid] = bi;
        __syncthreads();
        for (int s = 128; s > 0; s >>= 1) {
            if (tid < s) {
                if (rv[tid + s] > rv[tid] || (rv[tid + s] == rv[tid] && ri[tid + s] < ri[tid])) {
                    rv[tid] = rv[tid + s]; ri[tid] = ri[tid + s];
                }
            }
            __syncthreads();
        }
        if (tid == 0) { idx[b * KP + i] = ri[0]; row[ri[0]] = -1e30f; }
        __syncthreads();
    }
    if (tid < KP - KSEL) idx[b * KP + KSEL + tid] = 0;
}

// ---------------------------------------------------------------------------
// Kernel 2: gather + L2 normalize (f32 residual) + LayerNorm -> bf16 Q input
// ---------------------------------------------------------------------------
__global__ __launch_bounds__(256) void gather_kernel(const float* __restrict__ features,
                                                     const int* __restrict__ idx,
                                                     const float* __restrict__ lnw,
                                                     const float* __restrict__ lnb,
                                                     float* __restrict__ tokF,
                                                     __bf16* __restrict__ qB) {
    int rid = blockIdx.x;
    int b = rid / KP, t = rid - b * KP;
    int tid = threadIdx.x;
    float* trow = tokF + (size_t)rid * DDIM;
    __bf16* qrow = qB + (size_t)rid * DDIM;
    __shared__ float red[256];
    __shared__ float sc0, sc1;

    if (t >= KSEL) {
        for (int c = tid; c < DDIM; c += 256) { trow[c] = 0.f; qrow[c] = (__bf16)0.f; }
        return;
    }
    int j = idx[rid];
    const float* src = features + (size_t)b * NT * DDIM + (size_t)j * DDIM;
    float x0 = src[tid], x1 = src[tid + 256];

    red[tid] = x0 * x0 + x1 * x1;
    __syncthreads();
    for (int s = 128; s > 0; s >>= 1) { if (tid < s) red[tid] += red[tid + s]; __syncthreads(); }
    if (tid == 0) sc0 = 1.f / (sqrtf(red[0]) + 1e-6f);
    __syncthreads();
    float n = sc0;
    float y0 = x0 * n, y1 = x1 * n;
    trow[tid] = y0; trow[tid + 256] = y1;

    __syncthreads();
    red[tid] = y0 + y1;
    __syncthreads();
    for (int s = 128; s > 0; s >>= 1) { if (tid < s) red[tid] += red[tid + s]; __syncthreads(); }
    if (tid == 0) sc1 = red[0] / (float)DDIM;
    __syncthreads();
    float mean = sc1;
    float d0 = y0 - mean, d1 = y1 - mean;
    __syncthreads();
    red[tid] = d0 * d0 + d1 * d1;
    __syncthreads();
    for (int s = 128; s > 0; s >>= 1) { if (tid < s) red[tid] += red[tid + s]; __syncthreads(); }
    if (tid == 0) sc0 = rsqrtf(red[0] / (float)DDIM + 1e-5f);
    __syncthreads();
    float inv = sc0;
    qrow[tid]       = (__bf16)(d0 * inv * lnw[tid] + lnb[tid]);
    qrow[tid + 256] = (__bf16)(d1 * inv * lnw[tid + 256] + lnb[tid + 256]);
}

// ---------------------------------------------------------------------------
// Kernel 3: f32 -> bf16 weight conversion
// ---------------------------------------------------------------------------
__global__ void cvt_kernel(const float* __restrict__ s, __bf16* __restrict__ d, int n) {
    int i = blockIdx.x * blockDim.x + threadIdx.x;
    if (i < n) d[i] = (__bf16)s[i];
}

// ---------------------------------------------------------------------------
// Kernel 4: double-buffered async WMMA GEMM
//   C[M,N] = A[M,K] @ W[N,K]^T + bias, fused epilogue.
// Block 256 threads (8 waves), tile 128(M) x 128(N), K-step 32, unrolled x2
// so the LDS stage is a compile-time constant (no cndmask address selects).
// Wave tile 32x64: 8 wmma per K-step. Next K tile prefetched with
// GLOBAL_LOAD_ASYNC_TO_LDS_B128 into the other LDS stage while computing.
// mode 0: store bf16            mode 1: GELU -> bf16
// mode 2: tokF + sigmoid(g)*v -> bf16 (residual)
// mode 3: store f32             mode 4: + capF, masked atomic max pool -> fout
// ---------------------------------------------------------------------------
__global__ __launch_bounds__(256) void gemm_wmma_kernel(const __bf16* __restrict__ A,
                                                        const __bf16* __restrict__ W,
                                                        const float* __restrict__ bias,
                                                        int M, int N, int K, int mode,
                                                        __bf16* __restrict__ outB,
                                                        float* __restrict__ outF,
                                                        const float* __restrict__ resid,
                                                        const float* __restrict__ addF,
                                                        const float* __restrict__ gp,
                                                        const int* __restrict__ lens,
                                                        float* __restrict__ fout) {
    extern __shared__ __bf16 lds[];   // no static __shared__ here: base offset 0
    int tid = threadIdx.x, lane = tid & 31, wid = tid >> 5;
    int wm = wid >> 1, wn = wid & 1;
    int bm = blockIdx.y * 128, bn = blockIdx.x * 128;

    // per-thread async-copy assignment: 128x32 tile, 16 elems (2 x b128) each
    int rl = tid >> 1, cl = (tid & 1) * 16;
    const __bf16* pA = A + (size_t)(bm + rl) * K + cl;
    const __bf16* pW = W + (size_t)(bn + rl) * K + cl;
    const unsigned a0off = (unsigned)((rl * 40 + cl) * 2);
    const unsigned w0off = a0off + WS_BYTEOFF;

    v8f acc[2][4] = {};
    const int NKT = K >> 5;   // always even here (16 or 32)

    auto prefetch = [&](int s) {
        unsigned ao = a0off + (unsigned)s * STG_BYTES;
        unsigned wo = w0off + (unsigned)s * STG_BYTES;
        async_ld_b128(ao,      pA);
        async_ld_b128(ao + 16, pA + 8);
        async_ld_b128(wo,      pW);
        async_ld_b128(wo + 16, pW + 8);
        pA += 32; pW += 32;
    };
    auto compute = [&](int s) {
        const __bf16* Ab = lds + s * (STG_BYTES / 2);
        const __bf16* Wb = Ab + (WS_BYTEOFF / 2);
        v16bf a0 = ldfrag(&Ab[(wm * 32) * 40], 40, lane);
        v16bf a1 = ldfrag(&Ab[(wm * 32 + 16) * 40], 40, lane);
        v16bf b0 = ldfrag(&Wb[(wn * 64) * 40], 40, lane);
        v16bf b1 = ldfrag(&Wb[(wn * 64 + 16) * 40], 40, lane);
        v16bf b2 = ldfrag(&Wb[(wn * 64 + 32) * 40], 40, lane);
        v16bf b3 = ldfrag(&Wb[(wn * 64 + 48) * 40], 40, lane);
        acc[0][0] = wmma_bf16(a0, b0, acc[0][0]);
        acc[0][1] = wmma_bf16(a0, b1, acc[0][1]);
        acc[0][2] = wmma_bf16(a0, b2, acc[0][2]);
        acc[0][3] = wmma_bf16(a0, b3, acc[0][3]);
        acc[1][0] = wmma_bf16(a1, b0, acc[1][0]);
        acc[1][1] = wmma_bf16(a1, b1, acc[1][1]);
        acc[1][2] = wmma_bf16(a1, b2, acc[1][2]);
        acc[1][3] = wmma_bf16(a1, b3, acc[1][3]);
    };

    // prologue: async-prefetch K tile 0 into stage 0
    prefetch(0);
    wait_async0();
    __syncthreads();

    for (int kt = 0; kt < NKT; kt += 2) {
        // half A: compute stage 0 (tile kt); tile kt+1 always exists (NKT even)
        prefetch(1);
        compute(0);
        wait_async0();
        __syncthreads();
        // half B: compute stage 1 (tile kt+1); prefetch tile kt+2 if it exists
        if (kt + 2 < NKT) prefetch(0);
        compute(1);
        wait_async0();
        __syncthreads();
    }

    float sg = (mode == 2) ? 1.f / (1.f + __expf(-gp[0])) : 0.f;
    for (int sm = 0; sm < 2; sm++) {
        for (int sn = 0; sn < 4; sn++) {
            int rbase = bm + wm * 32 + sm * 16;
            int cbase = bn + wn * 64 + sn * 16;
            for (int v = 0; v < 8; v++) {
                int r = rbase + v + ((lane >> 4) << 3);
                int c = cbase + (lane & 15);
                float val = acc[sm][sn][v] + bias[c];
                size_t o = (size_t)r * N + c;
                if (mode == 0) {
                    outB[o] = (__bf16)val;
                } else if (mode == 1) {
                    float gel = 0.5f * val * (1.f + erff(val * 0.70710678f));
                    outB[o] = (__bf16)gel;
                } else if (mode == 2) {
                    outB[o] = (__bf16)(resid[o] + sg * val);
                } else if (mode == 3) {
                    outF[o] = val;
                } else {
                    val += addF[o];
                    int bb = r / KP, tt = r - bb * KP;
                    if (tt < lens[bb]) {
                        (void)__hip_atomic_fetch_max(&fout[(size_t)bb * N + c], val,
                                                     __ATOMIC_RELAXED, __HIP_MEMORY_SCOPE_AGENT);
                    }
                }
            }
        }
    }
}

// ---------------------------------------------------------------------------
// Kernel 5: attention for one (b, h, 16-query-row block); one wave per block.
// LDS < 64KB. qkv rows are [q(512) | k(512) | v(512)] bf16, row stride 1536.
// ---------------------------------------------------------------------------
__global__ __launch_bounds__(32) void attn_kernel(const __bf16* __restrict__ qkv,
                                                  __bf16* __restrict__ saB) {
    int b = blockIdx.x, h = blockIdx.y, rb = blockIdx.z;   // rb: 0..9
    int lane = threadIdx.x;
    __shared__ __bf16 Qt[16 * 72];
    __shared__ __bf16 Kt[KP * 72];
    __shared__ __bf16 Vt[64 * 168];    // transposed: [hd][t]
    __shared__ float  Ssm[16 * KP];
    __shared__ __bf16 Psm[16 * KP];

    const __bf16* base = qkv + (size_t)b * KP * TDIM;
    for (int i = lane; i < KP * 64; i += 32) {
        int t = i >> 6, c = i & 63;
        Kt[t * 72 + c]  = base[t * TDIM + DDIM + h * 64 + c];
        Vt[c * 168 + t] = base[t * TDIM + 2 * DDIM + h * 64 + c];
    }
    for (int i = lane; i < 16 * 64; i += 32) {
        int t = i >> 6, c = i & 63;
        Qt[t * 72 + c] = base[(rb * 16 + t) * TDIM + h * 64 + c];
    }
    __syncthreads();

    // scores = Q K^T / 8
    v16bf q0 = ldfrag(&Qt[0], 72, lane);
    v16bf q1 = ldfrag(&Qt[32], 72, lane);
    for (int j = 0; j < 10; j++) {
        v16bf k0 = ldfrag(&Kt[(j * 16) * 72], 72, lane);
        v16bf k1 = ldfrag(&Kt[(j * 16) * 72 + 32], 72, lane);
        v8f c = {};
        c = wmma_bf16(q0, k0, c);
        c = wmma_bf16(q1, k1, c);
        for (int v = 0; v < 8; v++) {
            int m = v + ((lane >> 4) << 3);
            int col = j * 16 + (lane & 15);
            Ssm[m * KP + col] = c[v] * 0.125f;
        }
    }
    __syncthreads();

    // masked softmax over cols < 153; lane pair per row
    {
        int rr = lane >> 1, hh = lane & 1;
        int c0 = hh * 80, c1 = c0 + 80;
        float mx = -1e30f;
        for (int cc = c0; cc < c1; cc++)
            if (cc < KSEL) mx = fmaxf(mx, Ssm[rr * KP + cc]);
        mx = fmaxf(mx, __shfl_xor(mx, 1, 32));
        float sum = 0.f;
        for (int cc = c0; cc < c1; cc++) {
            float e = (cc < KSEL) ? __expf(Ssm[rr * KP + cc] - mx) : 0.f;
            sum += e;
            Ssm[rr * KP + cc] = e;
        }
        sum += __shfl_xor(sum, 1, 32);
        float inv = 1.f / sum;
        for (int cc = c0; cc < c1; cc++)
            Psm[rr * KP + cc] = (__bf16)(Ssm[rr * KP + cc] * inv);
    }
    __syncthreads();

    // sa = P V : [16,160] @ [160,64]
    for (int nt = 0; nt < 4; nt++) {
        v8f c = {};
        for (int kt = 0; kt < 5; kt++) {
            v16bf a  = ldfrag(&Psm[kt * 32], KP, lane);
            v16bf bb = ldfrag(&Vt[(nt * 16) * 168 + kt * 32], 168, lane);
            c = wmma_bf16(a, bb, c);
        }
        for (int v = 0; v < 8; v++) {
            int m = v + ((lane >> 4) << 3);
            int col = h * 64 + nt * 16 + (lane & 15);
            saB[((size_t)b * KP + rb * 16 + m) * DDIM + col] = (__bf16)c[v];
        }
    }
}

// ---------------------------------------------------------------------------
// Kernel 6: BN stats over valid rows (t < 153): per-column scale/shift
// ---------------------------------------------------------------------------
__global__ __launch_bounds__(256) void bn_stats_kernel(const float* __restrict__ hF,
                                                       const float* __restrict__ bnw,
                                                       const float* __restrict__ bnb,
                                                       float* __restrict__ scale,
                                                       float* __restrict__ shift) {
    int jc = blockIdx.x, tid = threadIdx.x;
    __shared__ float s1[256], s2[256];
    float a = 0.f, q = 0.f;
    for (int r = tid; r < MR; r += 256) {
        int tt = r % KP;
        if (tt < KSEL) {
            float v = hF[(size_t)r * DDIM + jc];
            a += v; q += v * v;
        }
    }
    s1[tid] = a; s2[tid] = q;
    __syncthreads();
    for (int s = 128; s > 0; s >>= 1) {
        if (tid < s) { s1[tid] += s1[tid + s]; s2[tid] += s2[tid + s]; }
        __syncthreads();
    }
    if (tid == 0) {
        const float cnt = (float)(BATCH * KSEL);
        float mu = s1[0] / cnt;
        float var = s2[0] / cnt - mu * mu;
        float sc = bnw[jc] * rsqrtf(var + 1e-5f);
        scale[jc] = sc;
        shift[jc] = bnb[jc] - mu * sc;
    }
}

// ---------------------------------------------------------------------------
// Kernel 7: BN apply + ReLU -> bf16
// ---------------------------------------------------------------------------
__global__ void bn_apply_kernel(const float* __restrict__ hF,
                                const float* __restrict__ scale,
                                const float* __restrict__ shift,
                                __bf16* __restrict__ hb) {
    size_t i = (size_t)blockIdx.x * blockDim.x + threadIdx.x;
    if (i < (size_t)MR * DDIM) {
        int c = (int)(i & (DDIM - 1));
        float v = hF[i] * scale[c] + shift[c];
        hb[i] = (__bf16)fmaxf(v, 0.f);
    }
}

// ---------------------------------------------------------------------------
// Host: orchestration
// ---------------------------------------------------------------------------
extern "C" void kernel_launch(void* const* d_in, const int* in_sizes, int n_in,
                              void* d_out, int out_size, void* d_ws, size_t ws_size,
                              hipStream_t stream) {
    const float* features = (const float*)d_in[0];
    const int*   text     = (const int*)d_in[1];
    const float* atten    = (const float*)d_in[2];
    const float* ln_q_w   = (const float*)d_in[3];
    const float* ln_q_b   = (const float*)d_in[4];
    const float* sa_in_w  = (const float*)d_in[5];
    const float* sa_in_b  = (const float*)d_in[6];
    const float* sa_out_w = (const float*)d_in[7];
    const float* sa_out_b = (const float*)d_in[8];
    const float* ref_w1   = (const float*)d_in[9];
    const float* ref_b1   = (const float*)d_in[10];
    const float* ref_w2   = (const float*)d_in[11];
    const float* ref_b2   = (const float*)d_in[12];
    const float* gscalar  = (const float*)d_in[13];
    const float* lin_w    = (const float*)d_in[14];
    const float* lin_b    = (const float*)d_in[15];
    const float* mlp_w1   = (const float*)d_in[16];
    const float* mlp_b1   = (const float*)d_in[17];
    const float* bn_w     = (const float*)d_in[18];
    const float* bn_b     = (const float*)d_in[19];
    const float* mlp_w2   = (const float*)d_in[20];
    const float* mlp_b2   = (const float*)d_in[21];
    float* out = (float*)d_out;

    // ----- workspace layout (bump allocator) -----
    char* ws = (char*)d_ws;
    size_t o = 0;
    auto take = [&](size_t bytes) { char* p = ws + o; o += (bytes + 255) & ~(size_t)255; return p; };
    float*  tokF  = (float*)take((size_t)MR * DDIM * 4);
    __bf16* qB    = (__bf16*)take((size_t)MR * DDIM * 2);
    __bf16* qkvB  = (__bf16*)take((size_t)MR * TDIM * 2);
    __bf16* saB   = (__bf16*)take((size_t)MR * DDIM * 2);
    __bf16* sa2B  = (__bf16*)take((size_t)MR * DDIM * 2);
    __bf16* t1B   = (__bf16*)take((size_t)MR * EDIM * 2);
    __bf16* tokbB = (__bf16*)take((size_t)MR * DDIM * 2);
    float*  capF  = (float*)take((size_t)MR * EDIM * 4);
    float*  hF    = (float*)take((size_t)MR * DDIM * 4);
    __bf16* hbB   = (__bf16*)take((size_t)MR * DDIM * 2);
    __bf16* w_sain  = (__bf16*)take((size_t)TDIM * DDIM * 2);
    __bf16* w_saout = (__bf16*)take((size_t)DDIM * DDIM * 2);
    __bf16* w_r1    = (__bf16*)take((size_t)EDIM * DDIM * 2);
    __bf16* w_r2    = (__bf16*)take((size_t)DDIM * EDIM * 2);
    __bf16* w_lin   = (__bf16*)take((size_t)EDIM * DDIM * 2);
    __bf16* w_m1    = (__bf16*)take((size_t)DDIM * DDIM * 2);
    __bf16* w_m2    = (__bf16*)take((size_t)EDIM * DDIM * 2);
    int*   idx  = (int*)take((size_t)MR * 4);
    int*   lens = (int*)take((size_t)BATCH * 4);
    float* bnsc = (float*)take((size_t)DDIM * 4);
    float* bnsh = (float*)take((size_t)DDIM * 4);

    // ----- 0: init output for max pooling -----
    init_out_kernel<<<(BATCH * EDIM + 255) / 256, 256, 0, stream>>>(out, BATCH * EDIM);

    // ----- 1: top-k selection -----
    topk_kernel<<<BATCH, 256, 0, stream>>>(text, atten, idx, lens);

    // ----- 2: gather + normalize + layernorm -----
    gather_kernel<<<MR, 256, 0, stream>>>(features, idx, ln_q_w, ln_q_b, tokF, qB);

    // ----- 3: weight conversion to bf16 -----
    auto cvt = [&](const float* s, __bf16* d, int n) {
        cvt_kernel<<<(n + 255) / 256, 256, 0, stream>>>(s, d, n);
    };
    cvt(sa_in_w,  w_sain,  TDIM * DDIM);
    cvt(sa_out_w, w_saout, DDIM * DDIM);
    cvt(ref_w1,   w_r1,    EDIM * DDIM);
    cvt(ref_w2,   w_r2,    DDIM * EDIM);
    cvt(lin_w,    w_lin,   EDIM * DDIM);
    cvt(mlp_w1,   w_m1,    DDIM * DDIM);
    cvt(mlp_w2,   w_m2,    EDIM * DDIM);

    dim3 blk(256);
    // ----- G1: qkv = LN(tok) @ sa_in_w^T + b  [MR,1536] -----
    gemm_wmma_kernel<<<dim3(TDIM / 128, MR / 128), blk, GEMM_LDS, stream>>>(
        qB, w_sain, sa_in_b, MR, TDIM, DDIM, 0, qkvB, nullptr, nullptr, nullptr, nullptr, nullptr, nullptr);

    // ----- attention (8 heads, 10 row blocks, 256 batches) -----
    attn_kernel<<<dim3(BATCH, 8, KP / 16), 32, 0, stream>>>(qkvB, saB);

    // ----- G2: out-proj -----
    gemm_wmma_kernel<<<dim3(DDIM / 128, MR / 128), blk, GEMM_LDS, stream>>>(
        saB, w_saout, sa_out_b, MR, DDIM, DDIM, 0, sa2B, nullptr, nullptr, nullptr, nullptr, nullptr, nullptr);

    // ----- G3: refiner MLP up + GELU -----
    gemm_wmma_kernel<<<dim3(EDIM / 128, MR / 128), blk, GEMM_LDS, stream>>>(
        sa2B, w_r1, ref_b1, MR, EDIM, DDIM, 1, t1B, nullptr, nullptr, nullptr, nullptr, nullptr, nullptr);

    // ----- G4: refiner MLP down + sigmoid(g)-gated residual -> tok' (bf16) -----
    gemm_wmma_kernel<<<dim3(DDIM / 128, MR / 128), blk, GEMM_LDS, stream>>>(
        t1B, w_r2, ref_b2, MR, DDIM, EDIM, 2, tokbB, nullptr, tokF, nullptr, gscalar, nullptr, nullptr);

    // ----- G5: cap = tok' @ lin_w^T + b (f32) -----
    gemm_wmma_kernel<<<dim3(EDIM / 128, MR / 128), blk, GEMM_LDS, stream>>>(
        tokbB, w_lin, lin_b, MR, EDIM, DDIM, 3, nullptr, capF, nullptr, nullptr, nullptr, nullptr, nullptr);

    // ----- G6: h_pre = tok' @ mlp_w1^T + b (f32 for BN stats) -----
    gemm_wmma_kernel<<<dim3(DDIM / 128, MR / 128), blk, GEMM_LDS, stream>>>(
        tokbB, w_m1, mlp_b1, MR, DDIM, DDIM, 3, nullptr, hF, nullptr, nullptr, nullptr, nullptr, nullptr);

    // ----- BN (train-mode stats over valid rows) + ReLU -----
    bn_stats_kernel<<<DDIM, 256, 0, stream>>>(hF, bn_w, bn_b, bnsc, bnsh);
    bn_apply_kernel<<<(int)(((size_t)MR * DDIM + 255) / 256), 256, 0, stream>>>(hF, bnsc, bnsh, hbB);

    // ----- G7: feat = h @ mlp_w2^T + b + cap; masked max-pool into out -----
    gemm_wmma_kernel<<<dim3(EDIM / 128, MR / 128), blk, GEMM_LDS, stream>>>(
        hbB, w_m2, mlp_b2, MR, EDIM, DDIM, 4, nullptr, nullptr, nullptr, capF, nullptr, lens, out);

    (void)in_sizes; (void)n_in; (void)out_size; (void)ws_size;
}
